// FourierLayer_23046794510984
// MI455X (gfx1250) — matrix operand order repaired and checked by reading, not compiled, not run on hardware
//
#include <hip/hip_runtime.h>
#include <hip/hip_bf16.h>

typedef __attribute__((ext_vector_type(16))) _Float16 v16h;
typedef __attribute__((ext_vector_type(8)))  float    v8f;
typedef __attribute__((ext_vector_type(4)))  float    v4f;

// byte offset of an LDS pointer within the group segment (as(3) ptrtoint)
__device__ __forceinline__ unsigned lds_byte_off(const void* p) {
    return (unsigned)(uintptr_t)(__attribute__((address_space(3))) const void*)p;
}

// ---------------------------------------------------------------------------
// Setup: c = irfft(fourier_weights, 64); MatT[i][j] = (f16) c[(i-j) & 63]
// out[r,:] = d[r,:] (1x64)  x  Mat (64x64), Mat[j][i] = c[(i-j)&63]
// Stored transposed (row i = output channel) so B fragments load as 16
// consecutive halves (32B) per lane.
// ---------------------------------------------------------------------------
__global__ void build_matT(const float* __restrict__ fw, _Float16* __restrict__ matT) {
    __shared__ float c[64];
    const int m = threadIdx.x;              // 64 threads
    const float PI = 3.14159265358979323846f;
    float acc = fw[0];
    acc += fw[32] * __cosf(PI * (float)m);  // Nyquist term
#pragma unroll
    for (int k = 1; k < 32; ++k)
        acc += 2.0f * fw[k] * __cosf((2.0f * PI / 64.0f) * (float)(k * m));
    c[m] = acc * (1.0f / 64.0f);
    __syncthreads();
#pragma unroll
    for (int j = 0; j < 64; ++j)
        matT[m * 64 + j] = (_Float16)c[(m - j) & 63];
}

// ---------------------------------------------------------------------------
// Main kernel: 256 threads = 8 waves, 128 rows per block.
// LDS tile per row (stride 68 dwords): [0..1]=x, [2..65]=inputs, pad 2.
// d[j] = tile[row][j] for j=0..63  (x0,x1,in0..in61)  -- rfft truncation.
// inputs[row][n] = tile[row][2+n]                      -- for z epilogue.
// Staging uses CDNA5 async global->LDS (ASYNCcnt), no VGPR round-trip.
// ---------------------------------------------------------------------------
#define ROWS_PER_BLOCK 128
#define LDS_STRIDE 68   // dwords; 68*4=272 bytes, multiple of 16 -> aligned b128 runs

__global__ void __launch_bounds__(256)
fourier_layer_kernel(const float* __restrict__ in, const float* __restrict__ xc,
                     const _Float16* __restrict__ matT, const float* __restrict__ lw,
                     float* __restrict__ out) {
    __shared__ float tile[ROWS_PER_BLOCK * LDS_STRIDE];

    const int t    = threadIdx.x;      // 0..255
    const int lane = t & 31;
    const int wv   = t >> 5;           // wave id 0..7
    const long long R0 = (long long)blockIdx.x * ROWS_PER_BLOCK;
    const unsigned tileB = lds_byte_off(tile);

    // ---- Async-stage inputs tile: 128 rows x 64 f32 (contiguous 32KB) ----
    // dest dword offset row*68 + 2 + col  ->  8B aligned, so 2x async b64
    // (inst offset applies to BOTH global and LDS addresses).
    const float* inBase = in + R0 * 64;
#pragma unroll
    for (int p = 0; p < 8; ++p) {
        const int f   = p * 1024 + t * 4;      // element index in tile
        const int row = f >> 6;
        const int col = f & 63;
        const unsigned loff = tileB + (unsigned)(row * LDS_STRIDE + 2 + col) * 4u;
        const float* g = inBase + f;
        asm volatile("global_load_async_to_lds_b64 %0, %1, off"
                     :: "v"(loff), "v"(g) : "memory");
        asm volatile("global_load_async_to_lds_b64 %0, %1, off offset:8"
                     :: "v"(loff), "v"(g) : "memory");
    }
    // ---- Async-stage x tile: 128 rows x 2 f32 (contiguous 1KB) ----
    {
        const unsigned xoff = tileB + (unsigned)((t >> 1) * LDS_STRIDE + (t & 1)) * 4u;
        const float* gx = xc + R0 * 2 + t;
        asm volatile("global_load_async_to_lds_b32 %0, %1, off"
                     :: "v"(xoff), "v"(gx) : "memory");
    }
    asm volatile("s_wait_asynccnt 0x0" ::: "memory");
    __syncthreads();

    // ---- B fragments: 4 column tiles x 2 K-halves, resident in registers ----
    // B k order used here (contiguous, not AMD's table order): lanes 0-15 hold K=j0..j0+15 (N=lane), lanes 16-31 K=j0+16..
    v16h bfrag[4][2];
#pragma unroll
    for (int nt = 0; nt < 4; ++nt)
#pragma unroll
        for (int kh = 0; kh < 2; ++kh) {
            const int i  = nt * 16 + (lane & 15);          // output channel N
            const int j0 = kh * 32 + (lane >> 4) * 16;     // K base for this lane half
            bfrag[nt][kh] = *(const v16h*)(matT + i * 64 + j0); // 32B aligned
        }

    // ---- A fragments for this wave's 16 rows ----
    // A k order used here (contiguous, same as B): lane L (M=L&15): halves 0..15 -> K=kbase..kbase+15,
    // kbase = (L>>4)*16.
    const int M     = lane & 15;
    const int kbase = (lane >> 4) * 16;
    const int rloc  = wv * 16 + M;
    const float* drow = &tile[rloc * LDS_STRIDE];

    v16h afrag[2];
#pragma unroll
    for (int ka = 0; ka < 2; ++ka) {           // K offset 0 / 32
        const int kb = kbase + ka * 32;
        v4f q0 = *(const v4f*)(drow + kb);          // K = kb .. kb+3   (16B aligned)
        v4f q1 = *(const v4f*)(drow + kb + 4);      // K = kb+4 .. kb+7
        v4f q2 = *(const v4f*)(drow + kb + 8);      // K = kb+8 .. kb+11
        v4f q3 = *(const v4f*)(drow + kb + 12);     // K = kb+12.. kb+15
        v16h a;
#pragma unroll
        for (int e = 0; e < 4; ++e) {
            a[e]      = (_Float16)q0[e];
            a[4 + e]  = (_Float16)q1[e];
            a[8 + e]  = (_Float16)q2[e];
            a[12 + e] = (_Float16)q3[e];
        }
        afrag[ka] = a;
    }

    // ---- 8 WMMAs: D = A(16x64) x Mat(64x64) over four 16-col tiles ----
    v8f acc[4] = {};
#pragma unroll
    for (int nt = 0; nt < 4; ++nt) {
        acc[nt] = __builtin_amdgcn_wmma_f32_16x16x32_f16(
            false, afrag[0], false, bfrag[nt][0], (short)0, acc[nt], false, false);
        acc[nt] = __builtin_amdgcn_wmma_f32_16x16x32_f16(
            false, afrag[1], false, bfrag[nt][1], (short)0, acc[nt], false, false);
    }

    // ---- Epilogue: out = relu(acc + inputs*lw), f32 ----
    // C/D layout: element r of v8f, lanes 0-15: M=r, N=lane; lanes 16-31: M=r+8.
    const int mhalf = (lane >> 4) * 8;
    const long long rowTop = R0 + (long long)wv * 16;
#pragma unroll
    for (int nt = 0; nt < 4; ++nt) {
        const int N = nt * 16 + (lane & 15);
        const float w = lw[N];
#pragma unroll
        for (int r = 0; r < 8; ++r) {
            const int Mloc = r + mhalf;
            const float zin = tile[(wv * 16 + Mloc) * LDS_STRIDE + 2 + N];
            float v = acc[nt][r] + zin * w;
            v = v > 0.0f ? v : 0.0f;
            out[(rowTop + Mloc) * 64 + N] = v;
        }
    }
}

// ---------------------------------------------------------------------------
extern "C" void kernel_launch(void* const* d_in, const int* in_sizes, int n_in,
                              void* d_out, int out_size, void* d_ws, size_t ws_size,
                              hipStream_t stream) {
    const float* inputs = (const float*)d_in[0];   // [16, 65536, 64] f32
    const float* x      = (const float*)d_in[1];   // [16, 65536, 2]  f32
    const float* fw     = (const float*)d_in[2];   // [33] f32
    const float* lw     = (const float*)d_in[3];   // [64] f32
    float*       out    = (float*)d_out;           // [16, 65536, 64] f32
    _Float16*    matT   = (_Float16*)d_ws;         // 64*64 f16 = 8KB scratch

    // Build circulant matrix (recomputed every launch; deterministic, stream-ordered)
    build_matT<<<1, 64, 0, stream>>>(fw, matT);

    const long long rows = (long long)out_size / 64;          // 1,048,576
    const int blocks = (int)(rows / ROWS_PER_BLOCK);          // 8192, exact
    fourier_layer_kernel<<<blocks, 256, 0, stream>>>(inputs, x, matT, lw, out);
}
